// GNNEncoder_33234456936891
// MI455X (gfx1250) — compile-verified, matrix operand order checked by reading, not run on hardware
//
#include <hip/hip_runtime.h>
#include <hip/hip_bf16.h>

// ---------- CDNA5 WMMA types ----------
typedef __attribute__((ext_vector_type(16))) __bf16 v16bf;
typedef __attribute__((ext_vector_type(8)))  __bf16 v8bf;
typedef __attribute__((ext_vector_type(8)))  float  v8f;

// ---------------------------------------------------------------------------
// Convert W [K=256][Ncol=256] f32 (row major) -> WbT [n][k] bf16 so that a
// B-fragment lane (column n, 8 consecutive k) is one contiguous 16B load.
// ---------------------------------------------------------------------------
__global__ void convW_k(const float* __restrict__ W, unsigned short* __restrict__ WbT) {
    const int n = blockIdx.x;     // output column
    const int k = threadIdx.x;    // K index
    __bf16 v = (__bf16)W[(size_t)k * 256 + n];
    ((__bf16*)WbT)[(size_t)n * 256 + k] = v;
}

// ---------------------------------------------------------------------------
// H[N,256] = X[N,256] @ W[256,256] using v_wmma_f32_16x16x32_bf16.
// Block = 256 threads (8 waves). 128 rows of X are converted to bf16 in LDS;
// the 64-column B panel for each column group is staged in LDS too. All four
// B fragments of a K-step are loaded before the 4-WMMA burst so ds_loads
// pipeline and the WMMAs issue back-to-back.
// ---------------------------------------------------------------------------
__global__ void gemm256_bf16(const float* __restrict__ X,
                             const unsigned short* __restrict__ WbT_us,
                             float* __restrict__ Hout, int N) {
    __shared__ __bf16 Alds[128 * 256];                       // 64 KB
    __shared__ __bf16 Blds[64 * 256];                        // 32 KB
    const __bf16* __restrict__ WbT = (const __bf16*)WbT_us;

    const int tid  = threadIdx.x;
    const int row0 = blockIdx.x * 128;

    // Stage A tile: coalesced f32 loads, convert to bf16 in LDS.
    for (int i = 0; i < 128; ++i) {
        int r = row0 + i;
        float v = (r < N) ? X[(size_t)r * 256 + tid] : 0.0f;
        Alds[i * 256 + tid] = (__bf16)v;
    }

    const int wave = tid >> 5;
    const int lane = tid & 31;
    const int m    = lane & 15;           // row (A) / col (B) within 16
    const int ksel = (lane >> 4) * 8;     // K sub-offset per lane half
    const int arow = (wave * 16 + m) * 256;

    for (int cg = 0; cg < 4; ++cg) {      // 4 column groups of 64
        // Stage B panel: 64 cols x 256 K bf16 = 32 KB, 16B chunks.
        __syncthreads();                  // prev cg's B reads done (and A store at cg==0)
        {
            const v8bf* src = (const v8bf*)(WbT + (size_t)cg * 64 * 256);
            v8bf*       dst = (v8bf*)Blds;
#pragma unroll
            for (int i = 0; i < 8; ++i)   // 2048 chunks / 256 threads
                dst[tid + i * 256] = src[tid + i * 256];
        }
        __syncthreads();

        v8f acc[4] = {};
        for (int ks = 0; ks < 8; ++ks) {  // K = 256 in steps of 32
            const __bf16* ap = &Alds[arow + ks * 32 + ksel];
            v8bf alo = *(const v8bf*)ap;
            v8bf ahi = *(const v8bf*)(ap + 16);
            v16bf a = __builtin_shufflevector(alo, ahi,
                        0,1,2,3,4,5,6,7,8,9,10,11,12,13,14,15);

            // Load all 4 B fragments first: 8 ds_load_b128 in flight, one wait.
            v16bf bfrag[4];
#pragma unroll
            for (int t = 0; t < 4; ++t) {
                const __bf16* bp = &Blds[(t * 16 + m) * 256 + ks * 32 + ksel];
                v8bf blo = *(const v8bf*)bp;
                v8bf bhi = *(const v8bf*)(bp + 16);
                bfrag[t] = __builtin_shufflevector(blo, bhi,
                             0,1,2,3,4,5,6,7,8,9,10,11,12,13,14,15);
            }
#pragma unroll
            for (int t = 0; t < 4; ++t) {
                acc[t] = __builtin_amdgcn_wmma_f32_16x16x32_bf16(
                            false, a, false, bfrag[t], (short)0, acc[t], false, false);
            }
        }
        // C layout: VGPR r -> row r (lanes 0-15) / row r+8 (lanes 16-31), col = lane&15
        const int rowb = row0 + wave * 16 + (lane >> 4) * 8;
        const int colb = cg * 64 + m;
#pragma unroll
        for (int t = 0; t < 4; ++t) {
#pragma unroll
            for (int r = 0; r < 8; ++r) {
                int row = rowb + r;
                if (row < N) Hout[(size_t)row * 256 + colb + t * 16] = acc[t][r];
            }
        }
    }
}

// ---------------------------------------------------------------------------
// Per-node attention scores es/ed + init emax/denom.
// ---------------------------------------------------------------------------
__global__ void gat_prep_k(const float* __restrict__ Hf,
                           const float* __restrict__ asrc,
                           const float* __restrict__ adst,
                           float* es, float* ed, float* emax, float* denom, int N) {
    int tid = blockIdx.x * blockDim.x + threadIdx.x;
    if (tid >= N * 4) return;
    int h = tid & 3, n = tid >> 2;
    const float4* hp = (const float4*)(Hf + (size_t)n * 256 + h * 64);
    const float4* as = (const float4*)(asrc + h * 64);
    const float4* ad = (const float4*)(adst + h * 64);
    float ssum = 0.f, dsum = 0.f;
#pragma unroll
    for (int i = 0; i < 16; ++i) {
        float4 hv = hp[i], a = as[i], d = ad[i];
        ssum += hv.x*a.x + hv.y*a.y + hv.z*a.z + hv.w*a.w;
        dsum += hv.x*d.x + hv.y*d.y + hv.z*d.z + hv.w*d.w;
    }
    es[tid] = ssum;
    ed[tid] = dsum;
    emax[tid]  = -3.402823466e+38f;   // -FLT_MAX
    denom[tid] = 0.f;
}

__global__ void zero_k(float* p, size_t count) {
    for (size_t i = blockIdx.x * (size_t)blockDim.x + threadIdx.x; i < count;
         i += (size_t)gridDim.x * blockDim.x)
        p[i] = 0.f;
}

__device__ __forceinline__ void atomicMaxF(float* addr, float val) {
    if (val >= 0.f) atomicMax((int*)addr, __float_as_int(val));
    else            atomicMin((unsigned int*)addr, __float_as_uint(val));
}

__device__ __forceinline__ void edge_sd(const int* __restrict__ ei, int E, int t,
                                        int& s, int& d) {
    if (t < E) { s = ei[t]; d = ei[E + t]; }
    else       { s = d = t - E; }          // self loop
}

// Pass 1: leaky-relu edge score + segment max (per dst, per head).
__global__ void edge_max_k(const int* __restrict__ ei,
                           const float* __restrict__ es, const float* __restrict__ ed,
                           float* emax, float* __restrict__ elr, int E, int T) {
    int t = blockIdx.x * 256 + threadIdx.x;
    if (t >= T) return;
    int s, d; edge_sd(ei, E, t, s, d);
#pragma unroll
    for (int h = 0; h < 4; ++h) {
        float e = es[s * 4 + h] + ed[d * 4 + h];
        float lr = e > 0.f ? e : 0.2f * e;
        elr[(size_t)t * 4 + h] = lr;
        atomicMaxF(&emax[d * 4 + h], lr);
    }
}

// Pass 2: exp(e - max) + segment sum.
__global__ void edge_sum_k(const int* __restrict__ ei, float* __restrict__ elr,
                           const float* __restrict__ emax, float* denom, int E, int T) {
    int t = blockIdx.x * 256 + threadIdx.x;
    if (t >= T) return;
    int s, d; edge_sd(ei, E, t, s, d); (void)s;
#pragma unroll
    for (int h = 0; h < 4; ++h) {
        float ee = __expf(elr[(size_t)t * 4 + h] - emax[d * 4 + h]);
        elr[(size_t)t * 4 + h] = ee;
        atomicAdd(&denom[d * 4 + h], ee);
    }
}

// Pass 3: wave32 per edge; coalesced gather of h[src], alpha-weighted scatter-add.
__global__ void edge_agg_k(const int* __restrict__ ei, const float* __restrict__ elr,
                           const float* __restrict__ denom, const float* __restrict__ Hf,
                           float* agg, int E, int T) {
    int wv   = (blockIdx.x * 256 + threadIdx.x) >> 5;
    int lane = threadIdx.x & 31;
    if (wv >= T) return;
    int s, d; edge_sd(ei, E, wv, s, d);
    float al[4];
#pragma unroll
    for (int h = 0; h < 4; ++h)
        al[h] = elr[(size_t)wv * 4 + h] / (denom[d * 4 + h] + 1e-16f);
    const float* hsrc = Hf + (size_t)s * 256;
    float* outp = agg + (size_t)d * 256;
    __builtin_prefetch(hsrc, 0, 0);        // global_prefetch_b8
#pragma unroll
    for (int it = 0; it < 8; ++it) {
        int f = it * 32 + lane;            // head = it>>1 (uniform per iteration)
        atomicAdd(&outp[f], al[it >> 1] * hsrc[f]);
    }
}

__global__ void relu_bias_k(float* A, const float* __restrict__ b, size_t count) {
    for (size_t i = blockIdx.x * (size_t)blockDim.x + threadIdx.x; i < count;
         i += (size_t)gridDim.x * blockDim.x) {
        float v = A[i] + b[i & 255];
        A[i] = v > 0.f ? v : 0.f;
    }
}

// emb = agg + b2 (to d_out), n_part[n] = emb[n,:] . Wc[256:512]  (wave per node)
__global__ void finalize_k(const float* __restrict__ agg, const float* __restrict__ b2,
                           const float* __restrict__ Wc, float* __restrict__ emb,
                           float* __restrict__ npart, int N) {
    int n    = (blockIdx.x * 256 + threadIdx.x) >> 5;
    int lane = threadIdx.x & 31;
    if (n >= N) return;
    const float* WcN = Wc + 256;
    float sum = 0.f;
#pragma unroll
    for (int it = 0; it < 8; ++it) {
        int f = it * 32 + lane;
        float v = agg[(size_t)n * 256 + f] + b2[f];
        emb[(size_t)n * 256 + f] = v;
        sum += v * WcN[f];
    }
#pragma unroll
    for (int off = 16; off; off >>= 1) sum += __shfl_down(sum, off, 32);
    if (lane == 0) npart[n] = sum;
}

// q_part[q] = query[q,:] . Wc[:256] + bc   (wave per query; one block)
__global__ void qpart_k(const float* __restrict__ query, const float* __restrict__ Wc,
                        const float* __restrict__ bc, float* qpart, int Q) {
    int q    = threadIdx.x >> 5;
    int lane = threadIdx.x & 31;
    if (q >= Q) return;
    float sum = 0.f;
#pragma unroll
    for (int it = 0; it < 8; ++it) {
        int f = it * 32 + lane;
        sum += query[(size_t)q * 256 + f] * Wc[f];
    }
#pragma unroll
    for (int off = 16; off; off >>= 1) sum += __shfl_down(sum, off, 32);
    if (lane == 0) qpart[q] = sum + bc[0];
}

__global__ void scores_k(const float* __restrict__ npart, const float* __restrict__ qpart,
                         float* __restrict__ out, int N, int Q) {
    int n = blockIdx.x * 256 + threadIdx.x;
    if (n >= N) return;
    float np = npart[n];
    for (int q = 0; q < Q; ++q) out[(size_t)q * N + n] = qpart[q] + np;
}

// ---------------------------------------------------------------------------
extern "C" void kernel_launch(void* const* d_in, const int* in_sizes, int n_in,
                              void* d_out, int out_size, void* d_ws, size_t ws_size,
                              hipStream_t stream) {
    const float* x      = (const float*)d_in[0];
    const int*   ei     = (const int*)  d_in[1];
    const float* query  = (const float*)d_in[2];
    const float* W1     = (const float*)d_in[3];
    const float* asrc1  = (const float*)d_in[4];
    const float* adst1  = (const float*)d_in[5];
    const float* b1     = (const float*)d_in[6];
    const float* W2     = (const float*)d_in[7];
    const float* asrc2  = (const float*)d_in[8];
    const float* adst2  = (const float*)d_in[9];
    const float* b2     = (const float*)d_in[10];
    const float* Wc     = (const float*)d_in[11];
    const float* bc     = (const float*)d_in[12];

    const int N = in_sizes[0] / 256;
    const int E = in_sizes[1] / 2;
    const int Q = in_sizes[2] / 256;
    const int T = E + N;                    // edges incl. self-loops

    // ---- workspace carve ----
    char* wp = (char*)d_ws;
    auto carve = [&](size_t bytes) -> char* {
        char* p = wp; wp += (bytes + 255) & ~(size_t)255; return p;
    };
    unsigned short* WbT1 = (unsigned short*)carve(256 * 256 * 2);
    unsigned short* WbT2 = (unsigned short*)carve(256 * 256 * 2);
    float* Hbuf  = (float*)carve((size_t)N * 256 * 4);   // h = X@W (per layer)
    float* AGG   = (float*)carve((size_t)N * 256 * 4);   // aggregation / relu output
    float* es    = (float*)carve((size_t)N * 4 * 4);
    float* ed    = (float*)carve((size_t)N * 4 * 4);
    float* emax  = (float*)carve((size_t)N * 4 * 4);
    float* denom = (float*)carve((size_t)N * 4 * 4);
    float* elr   = (float*)carve((size_t)T * 4 * 4);
    float* npart = (float*)carve((size_t)N * 4);
    float* qpart = (float*)carve(64);

    float* emb_out    = (float*)d_out;
    float* scores_out = emb_out + (size_t)N * 256;

    const int gemmBlocks = (N + 127) / 128;
    const int nhBlocks   = (N * 4 + 255) / 256;
    const int eBlocks    = (T + 255) / 256;
    const int eWBlocks   = (T + 7) / 8;      // wave per edge
    const int nWBlocks   = (N + 7) / 8;      // wave per node

    // Weight conversion (once per call; L2-resident afterwards)
    convW_k<<<256, 256, 0, stream>>>(W1, WbT1);
    convW_k<<<256, 256, 0, stream>>>(W2, WbT2);

    // ---------------- Layer 1 ----------------
    gemm256_bf16<<<gemmBlocks, 256, 0, stream>>>(x, WbT1, Hbuf, N);
    gat_prep_k<<<nhBlocks, 256, 0, stream>>>(Hbuf, asrc1, adst1, es, ed, emax, denom, N);
    zero_k<<<2048, 256, 0, stream>>>(AGG, (size_t)N * 256);
    edge_max_k<<<eBlocks, 256, 0, stream>>>(ei, es, ed, emax, elr, E, T);
    edge_sum_k<<<eBlocks, 256, 0, stream>>>(ei, elr, emax, denom, E, T);
    edge_agg_k<<<eWBlocks, 256, 0, stream>>>(ei, elr, denom, Hbuf, AGG, E, T);
    relu_bias_k<<<2048, 256, 0, stream>>>(AGG, b1, (size_t)N * 256);

    // ---------------- Layer 2 ----------------
    gemm256_bf16<<<gemmBlocks, 256, 0, stream>>>(AGG, WbT2, Hbuf, N);
    gat_prep_k<<<nhBlocks, 256, 0, stream>>>(Hbuf, asrc2, adst2, es, ed, emax, denom, N);
    zero_k<<<2048, 256, 0, stream>>>(AGG, (size_t)N * 256);   // after gemm read (stream order)
    edge_max_k<<<eBlocks, 256, 0, stream>>>(ei, es, ed, emax, elr, E, T);
    edge_sum_k<<<eBlocks, 256, 0, stream>>>(ei, elr, emax, denom, E, T);
    edge_agg_k<<<eWBlocks, 256, 0, stream>>>(ei, elr, denom, Hbuf, AGG, E, T);

    // ---------------- Output head ----------------
    finalize_k<<<nWBlocks, 256, 0, stream>>>(AGG, b2, Wc, emb_out, npart, N);
    qpart_k<<<1, 256, 0, stream>>>(query, Wc, bc, qpart, Q);
    scores_k<<<(N + 255) / 256, 256, 0, stream>>>(npart, qpart, scores_out, N, Q);
}
